// Torso_22110491640333
// MI455X (gfx1250) — matrix operand order repaired and checked by reading, not compiled, not run on hardware
//
#include <hip/hip_runtime.h>
#include <math.h>

// ---------------------------------------------------------------------------
// MI455X (gfx1250) implementation.
//  - Phase 1: bandwidth-bound GEMV over the 677MB g2g weight (float4 streaming,
//             gi vector staged in LDS, wave32 shuffle reduction).
//  - Phase 2: 24 sequential attention steps; all GEMMs use
//             v_wmma_f32_16x16x32_bf16 (bf16 operands, f32 accumulation).
//             Ping-pong buffers remove the read-column/write-row race.
// ---------------------------------------------------------------------------

typedef __bf16 bf16_t;
typedef __attribute__((ext_vector_type(16))) __bf16 v16bf;
typedef __attribute__((ext_vector_type(8)))  float  v8f;

#define S12   12
#define C32   32
#define L24   24          // L2 = 2*S
#define DH512 512
#define NHEAD 16
#define GIN   12240       // S*S*(T*S+1)
#define GROWS 4608        // S*S*C
#define A_LAYERS 8

__device__ __forceinline__ v8f wmma_bf16(v16bf a, v16bf b, v8f c) {
  return __builtin_amdgcn_wmma_f32_16x16x32_bf16(false, a, false, b,
                                                 (short)0, c, false, false);
}

// A fragment: 16x32 bf16, source is LDS row-major [M][ldk].
// Lane layout: lanes 0-15 m=lane, K=0..15;
// lanes 16-31 K=16..31.
__device__ __forceinline__ v16bf load_A(const bf16_t* base, int ldk,
                                        int mTile, int k0, int mValid) {
  int lane = threadIdx.x & 31;
  int m  = mTile * 16 + (lane & 15);
  int hi = lane >> 4;
  bool ok = (m < mValid);
  v16bf r;
#pragma unroll
  for (int v = 0; v < 8; ++v) {
    int kb = k0 + hi * 16 + 2 * v;
    r[2 * v]     = ok ? base[m * ldk + kb]     : (bf16_t)0.0f;
    r[2 * v + 1] = ok ? base[m * ldk + kb + 1] : (bf16_t)0.0f;
  }
  return r;
}

// B fragment: 32x16, B[k][n] = W[n][k] with k contiguous, W in GLOBAL f32.
// Lane n = tile*16 + (lane&15); lanes 0-15 hold K=0..15, 16-31 hold K=16..31.
__device__ __forceinline__ v16bf load_Bg(const float* base, int ldk,
                                         int nTile, int k0, int nValid) {
  int lane = threadIdx.x & 31;
  int n  = nTile * 16 + (lane & 15);
  int hi = lane >> 4;
  bool ok = (n < nValid);
  const float* p = base + (size_t)n * ldk + k0 + hi * 16;
  v16bf r;
#pragma unroll
  for (int v = 0; v < 8; ++v) {
    float e0 = ok ? p[2 * v]     : 0.0f;
    float e1 = ok ? p[2 * v + 1] : 0.0f;
    r[2 * v]     = (bf16_t)e0;
    r[2 * v + 1] = (bf16_t)e1;
  }
  return r;
}

// Same but source is LDS bf16 [N][ldk].
__device__ __forceinline__ v16bf load_Bl(const bf16_t* base, int ldk,
                                         int nTile, int k0, int nValid) {
  int lane = threadIdx.x & 31;
  int n  = nTile * 16 + (lane & 15);
  int hi = lane >> 4;
  bool ok = (n < nValid);
  v16bf r;
#pragma unroll
  for (int v = 0; v < 8; ++v) {
    int kb = k0 + hi * 16 + 2 * v;
    r[2 * v]     = ok ? base[n * ldk + kb]     : (bf16_t)0.0f;
    r[2 * v + 1] = ok ? base[n * ldk + kb + 1] : (bf16_t)0.0f;
  }
  return r;
}

// B fragment where memory layout is [K][ldn] (strided K), e.g. V in scores@V.
__device__ __forceinline__ v16bf load_BT(const bf16_t* base, int ldn,
                                         int nTile, int kValid) {
  int lane = threadIdx.x & 31;
  int n  = nTile * 16 + (lane & 15);
  int hi = lane >> 4;
  v16bf r;
#pragma unroll
  for (int v = 0; v < 8; ++v) {
    int k = hi * 16 + 2 * v;
    r[2 * v]     = (k     < kValid) ? base[k * ldn + n]       : (bf16_t)0.0f;
    r[2 * v + 1] = (k + 1 < kValid) ? base[(k + 1) * ldn + n] : (bf16_t)0.0f;
  }
  return r;
}

// D layout per ISA: lane n = tile*16+(lane&15); VGPR v -> row m = 8*hi + v.
__device__ __forceinline__ void storeD_bf16(bf16_t* dst, int ldn, v8f d,
                                            int mTile, int nTile, int mValid,
                                            float scale, float bias) {
  int lane = threadIdx.x & 31;
  int n  = nTile * 16 + (lane & 15);
  int hi = lane >> 4;
#pragma unroll
  for (int v = 0; v < 8; ++v) {
    int m = mTile * 16 + hi * 8 + v;
    if (m < mValid) dst[m * ldn + n] = (bf16_t)(d[v] * scale + bias);
  }
}

__device__ __forceinline__ void accD(float* dst, int ldn, v8f d,
                                     int mTile, int nTile, int mValid) {
  int lane = threadIdx.x & 31;
  int n  = nTile * 16 + (lane & 15);
  int hi = lane >> 4;
#pragma unroll
  for (int v = 0; v < 8; ++v) {
    int m = mTile * 16 + hi * 8 + v;
    if (m < mValid) atomicAdd(&dst[m * ldn + n], d[v]);   // ds_add_f32
  }
}

__device__ __forceinline__ void storeGelu(bf16_t* dst, int ldn, v8f d,
                                          int mTile, int nTile, int mValid,
                                          float bias) {
  int lane = threadIdx.x & 31;
  int n  = nTile * 16 + (lane & 15);
  int hi = lane >> 4;
#pragma unroll
  for (int v = 0; v < 8; ++v) {
    int m = mTile * 16 + hi * 8 + v;
    if (m < mValid) {
      float t = d[v] + bias;
      float g = 0.5f * t * (1.0f + erff(t * 0.70710678118654752f));
      dst[m * ldn + n] = (bf16_t)g;
    }
  }
}

// ---------------------------------------------------------------------------
// Kernel 1: build the three 12240-element gi vectors (transposed views of
// input_t + scalar projection appended every 85th element).
// ---------------------------------------------------------------------------
__global__ __launch_bounds__(256) void prep_kernel(
    const float* __restrict__ t, const float* __restrict__ sc,
    const float* __restrict__ s2gw, const float* __restrict__ s2gb,
    float* __restrict__ gi) {
  int idx = blockIdx.x * 256 + threadIdx.x;
  if (idx >= 3 * GIN) return;
  int i = idx / GIN, rem = idx % GIN;
  int ab = rem / 85, c = rem % 85;
  int a = ab / 12, b = ab % 12;
  float val;
  if (c < 84) {
    int s3 = c / 7, tt = c % 7;
    if (i == 0)      val = t[tt * 1728 + a  * 144 + b  * 12 + s3];
    else if (i == 1) val = t[tt * 1728 + b  * 144 + s3 * 12 + a ];
    else             val = t[tt * 1728 + s3 * 144 + a  * 12 + b ];
  } else {
    const float* wr = s2gw + (i * 144 + ab) * 8;
    float acc = s2gb[i * 144 + ab];
#pragma unroll
    for (int j = 0; j < 8; ++j) acc += wr[j] * sc[j];
    val = acc;
  }
  gi[idx] = val;
}

// ---------------------------------------------------------------------------
// Kernel 2: the 677MB GEMV. One block = 32 rows of one grid; gi staged in LDS.
// ---------------------------------------------------------------------------
__global__ __launch_bounds__(256) void gemv_kernel(
    const float* __restrict__ W, const float* __restrict__ bias,
    const float* __restrict__ gi, float* __restrict__ xout) {
  __shared__ __align__(16) float g[GIN];
  int i = blockIdx.x / 144;
  int rowBase = (blockIdx.x % 144) * 32;
  const float* gsrc = gi + i * GIN;
  for (int k = threadIdx.x; k < GIN; k += 256) g[k] = gsrc[k];
  __syncthreads();
  int wave = threadIdx.x >> 5, lane = threadIdx.x & 31;
  const float4* g4 = (const float4*)g;
  for (int rr = 0; rr < 4; ++rr) {
    int row = rowBase + wave * 4 + rr;
    const float4* w4 = (const float4*)(W + ((size_t)i * GROWS + row) * GIN);
    float acc = 0.0f;
    for (int k = lane; k < GIN / 4; k += 32) {
      float4 gv = g4[k];
      float4 wv = w4[k];
      acc = fmaf(gv.x, wv.x, fmaf(gv.y, wv.y,
            fmaf(gv.z, wv.z, fmaf(gv.w, wv.w, acc))));
    }
#pragma unroll
    for (int off = 16; off > 0; off >>= 1) acc += __shfl_xor(acc, off, 32);
    if (lane == 0) xout[i * GROWS + row] = acc + bias[i * GROWS + row];
  }
}

// ---------------------------------------------------------------------------
// Kernel 3: one attention step (layer, pair). blockIdx.x = s, 256 threads.
// ---------------------------------------------------------------------------
__device__ __forceinline__ void project_head(const bf16_t* actb,
                                             const float* W, const float* B,
                                             int h, bf16_t* dst) {
  v16bf a0 = load_A(actb, 32, 0, 0, L24);
  v16bf a1 = load_A(actb, 32, 1, 0, L24);
  const float* Wb = W + (size_t)(h * 32) * 32;
  v16bf b0 = load_Bg(Wb, 32, 0, 0, 32);
  v16bf b1 = load_Bg(Wb, 32, 1, 0, 32);
  v8f z = {0.f, 0.f, 0.f, 0.f, 0.f, 0.f, 0.f, 0.f};
  v8f d00 = wmma_bf16(a0, b0, z);
  v8f d01 = wmma_bf16(a0, b1, z);
  v8f d10 = wmma_bf16(a1, b0, z);
  v8f d11 = wmma_bf16(a1, b1, z);
  int lane = threadIdx.x & 31;
  float bb0 = B[h * 32 + (lane & 15)];
  float bb1 = B[h * 32 + 16 + (lane & 15)];
  storeD_bf16(dst, 32, d00, 0, 0, L24, 1.f, bb0);
  storeD_bf16(dst, 32, d01, 0, 1, L24, 1.f, bb1);
  storeD_bf16(dst, 32, d10, 1, 0, L24, 1.f, bb0);
  storeD_bf16(dst, 32, d11, 1, 1, L24, 1.f, bb1);
}

__global__ __launch_bounds__(256) void attn_step_kernel(
    const float* __restrict__ in1, const float* __restrict__ in2,
    float* __restrict__ out1, float* __restrict__ out2,
    const float* __restrict__ lnxg, const float* __restrict__ lnxb,
    const float* __restrict__ lnyg, const float* __restrict__ lnyb,
    const float* __restrict__ lnfg, const float* __restrict__ lnfb,
    const float* __restrict__ qw, const float* __restrict__ qb,
    const float* __restrict__ kw, const float* __restrict__ kb,
    const float* __restrict__ vw, const float* __restrict__ vb,
    const float* __restrict__ w1, const float* __restrict__ b1,
    const float* __restrict__ w2, const float* __restrict__ b2,
    const float* __restrict__ w3, const float* __restrict__ b3,
    int layer) {
  __shared__ float abuf[L24 * C32];
  __shared__ float xF[L24 * C32];
  __shared__ float red[256], red2[256];
  __shared__ float stats[2];
  __shared__ bf16_t xnb[L24 * C32], ynb[L24 * C32], hb[L24 * C32];
  __shared__ bf16_t gb[L24 * 128];
  __shared__ bf16_t pwQ[8][L24 * C32], pwK[8][L24 * C32], pwV[8][L24 * C32];

  int s = blockIdx.x;
  int tid = threadIdx.x;
  int wave = tid >> 5, lane = tid & 31;
  size_t ls = (size_t)layer * S12 + s;

  const float* LNXG = lnxg + ls * 768;  const float* LNXB = lnxb + ls * 768;
  const float* LNYG = lnyg + ls * 768;  const float* LNYB = lnyb + ls * 768;
  const float* LNFG = lnfg + ls * 768;  const float* LNFB = lnfb + ls * 768;
  const float* QW = qw + ls * 16384;    const float* QB = qb + ls * 512;
  const float* KW = kw + ls * 16384;    const float* KB = kb + ls * 512;
  const float* VW = vw + ls * 16384;    const float* VB = vb + ls * 512;
  const float* W1 = w1 + ls * 16384;    const float* B1 = b1 + ls * 32;
  const float* W2 = w2 + ls * 4096;     const float* B2 = b2 + ls * 128;
  const float* W3 = w3 + ls * 4096;     const float* B3 = b3 + ls * 32;

  // ---- load a = [x[m1][s,:,:]; x[m2][:,s,:]] ----
  for (int idx = tid; idx < 768; idx += 256) {
    int r = idx >> 5, c = idx & 31;
    abuf[idx] = (r < S12) ? in1[(s * S12 + r) * 32 + c]
                          : in2[((r - S12) * S12 + s) * 32 + c];
  }
  __syncthreads();

  // ---- whole-matrix LN stats of a ----
  {
    float s1 = 0.f, s2 = 0.f;
    for (int idx = tid; idx < 768; idx += 256) {
      float z = abuf[idx]; s1 += z; s2 += z * z;
    }
    red[tid] = s1; red2[tid] = s2;
    __syncthreads();
    for (int off = 128; off > 0; off >>= 1) {
      if (tid < off) { red[tid] += red[tid + off]; red2[tid] += red2[tid + off]; }
      __syncthreads();
    }
    if (tid == 0) {
      float mu = red[0] * (1.f / 768.f);
      float var = red2[0] * (1.f / 768.f) - mu * mu;
      stats[0] = mu; stats[1] = rsqrtf(var + 1e-5f);
    }
    __syncthreads();
  }
  {
    float mu = stats[0], rin = stats[1];
    for (int idx = tid; idx < 768; idx += 256) {
      float zn = (abuf[idx] - mu) * rin;
      xnb[idx] = (bf16_t)(zn * LNXG[idx] + LNXB[idx]);
      ynb[idx] = (bf16_t)(zn * LNYG[idx] + LNYB[idx]);
    }
  }
  // residual accumulator x = a + b1 (head contributions ds_add'ed in)
  for (int idx = tid; idx < 768; idx += 256)
    xF[idx] = abuf[idx] + B1[idx & 31];
  __syncthreads();

  // ---- per-wave head processing: wave w owns heads {w, w+8} ----
  bf16_t* Q = pwQ[wave];
  bf16_t* K = pwK[wave];
  bf16_t* V = pwV[wave];
  const v8f vz = {0.f, 0.f, 0.f, 0.f, 0.f, 0.f, 0.f, 0.f};
  for (int hh = 0; hh < 2; ++hh) {
    int h = wave + 8 * hh;
    project_head(xnb, QW, QB, h, Q);
    project_head(ynb, KW, KB, h, K);
    project_head(ynb, VW, VB, h, V);
    // scores = (q @ k^T) / sqrt(D)  -> stored into K (k consumed)
    {
      v16bf a0 = load_A(Q, 32, 0, 0, L24);
      v16bf a1 = load_A(Q, 32, 1, 0, L24);
      v16bf b0 = load_Bl(K, 32, 0, 0, L24);
      v16bf b1 = load_Bl(K, 32, 1, 0, L24);
      v8f d00 = wmma_bf16(a0, b0, vz);
      v8f d01 = wmma_bf16(a0, b1, vz);
      v8f d10 = wmma_bf16(a1, b0, vz);
      v8f d11 = wmma_bf16(a1, b1, vz);
      const float sc = 0.17677669529663688f;  // 1/sqrt(32)
      storeD_bf16(K, 32, d00, 0, 0, L24, sc, 0.f);
      storeD_bf16(K, 32, d01, 0, 1, L24, sc, 0.f);
      storeD_bf16(K, 32, d10, 1, 0, L24, sc, 0.f);
      storeD_bf16(K, 32, d11, 1, 1, L24, sc, 0.f);
    }
    // o = scores @ v  -> stored into Q (q consumed)
    {
      v16bf a0 = load_A(K, 32, 0, 0, L24);
      v16bf a1 = load_A(K, 32, 1, 0, L24);
      v16bf b0 = load_BT(V, 32, 0, L24);
      v16bf b1 = load_BT(V, 32, 1, L24);
      v8f d00 = wmma_bf16(a0, b0, vz);
      v8f d01 = wmma_bf16(a0, b1, vz);
      v8f d10 = wmma_bf16(a1, b0, vz);
      v8f d11 = wmma_bf16(a1, b1, vz);
      storeD_bf16(Q, 32, d00, 0, 0, L24, 1.f, 0.f);
      storeD_bf16(Q, 32, d01, 0, 1, L24, 1.f, 0.f);
      storeD_bf16(Q, 32, d10, 1, 0, L24, 1.f, 0.f);
      storeD_bf16(Q, 32, d11, 1, 1, L24, 1.f, 0.f);
    }
    // x += o_h @ w1[:, h*32:(h+1)*32]^T   (K-block of the 512-K GEMM)
    {
      v16bf a0 = load_A(Q, 32, 0, 0, L24);
      v16bf a1 = load_A(Q, 32, 1, 0, L24);
      const float* Wb = W1 + h * 32;     // w1[c][512] ; B[k=d][n=c]
      v16bf b0 = load_Bg(Wb, 512, 0, 0, 32);
      v16bf b1 = load_Bg(Wb, 512, 1, 0, 32);
      v8f d00 = wmma_bf16(a0, b0, vz);
      v8f d01 = wmma_bf16(a0, b1, vz);
      v8f d10 = wmma_bf16(a1, b0, vz);
      v8f d11 = wmma_bf16(a1, b1, vz);
      accD(xF, 32, d00, 0, 0, L24);
      accD(xF, 32, d01, 0, 1, L24);
      accD(xF, 32, d10, 1, 0, L24);
      accD(xF, 32, d11, 1, 1, L24);
    }
  }
  __syncthreads();

  // ---- LN(x) -> h ----
  {
    float s1 = 0.f, s2 = 0.f;
    for (int idx = tid; idx < 768; idx += 256) {
      float z = xF[idx]; s1 += z; s2 += z * z;
    }
    red[tid] = s1; red2[tid] = s2;
    __syncthreads();
    for (int off = 128; off > 0; off >>= 1) {
      if (tid < off) { red[tid] += red[tid + off]; red2[tid] += red2[tid + off]; }
      __syncthreads();
    }
    if (tid == 0) {
      float mu = red[0] * (1.f / 768.f);
      float var = red2[0] * (1.f / 768.f) - mu * mu;
      stats[0] = mu; stats[1] = rsqrtf(var + 1e-5f);
    }
    __syncthreads();
    float mu = stats[0], rin = stats[1];
    for (int idx = tid; idx < 768; idx += 256)
      hb[idx] = (bf16_t)((xF[idx] - mu) * rin * LNFG[idx] + LNFB[idx]);
  }
  __syncthreads();

  // ---- l2 + exact GeLU: (24,32)@(32,128); wave w owns n-tile w ----
  {
    int nT = wave;
    v16bf b0 = load_Bg(W2, 32, nT, 0, 128);
    v16bf a0 = load_A(hb, 32, 0, 0, L24);
    v16bf a1 = load_A(hb, 32, 1, 0, L24);
    v8f d0 = wmma_bf16(a0, b0, vz);
    v8f d1 = wmma_bf16(a1, b0, vz);
    float bb = B2[nT * 16 + (lane & 15)];
    storeGelu(gb, 128, d0, 0, nT, L24, bb);
    storeGelu(gb, 128, d1, 1, nT, L24, bb);
  }
  __syncthreads();

  // ---- l3 + residual + scatter: (24,128)@(128,32); waves 0-3, K-loop ----
  if (wave < 4) {
    int mT = wave >> 1, nT = wave & 1;
    v8f acc = vz;
    for (int kk = 0; kk < 4; ++kk) {
      v16bf a = load_A(gb, 128, mT, kk * 32, L24);
      v16bf b = load_Bg(W3, 128, nT, kk * 32, 32);
      acc = wmma_bf16(a, b, acc);
    }
    float bb = B3[nT * 16 + (lane & 15)];
    int n = nT * 16 + (lane & 15);
    int hi = lane >> 4;
#pragma unroll
    for (int v = 0; v < 8; ++v) {
      int m = mT * 16 + hi * 8 + v;
      if (m < L24) {
        float val = xF[m * 32 + n] + acc[v] + bb;
        if (m < S12) out1[(s * S12 + m) * 32 + n] = val;
        else         out2[(s * S12 + (m - S12)) * 32 + n] = val;
      }
    }
  }
}

// ---------------------------------------------------------------------------
// Kernel 4: final pack -> stack(x, axis=1).reshape(432, 32)
// ---------------------------------------------------------------------------
__global__ __launch_bounds__(256) void pack_kernel(
    const float* __restrict__ xbuf, int v0, int v1, int v2,
    float* __restrict__ out) {
  int idx = blockIdx.x * 256 + threadIdx.x;
  if (idx >= 432 * 32) return;
  int ch = idx & 31, row = idx >> 5;
  int a = row / 36, r = row % 36, i = r / 12, b = r % 12;
  int ver = (i == 0) ? v0 : ((i == 1) ? v1 : v2);
  out[idx] = xbuf[((size_t)ver * 3 + i) * GROWS + (a * 12 + b) * 32 + ch];
}

// ---------------------------------------------------------------------------
extern "C" void kernel_launch(void* const* d_in, const int* in_sizes, int n_in,
                              void* d_out, int out_size, void* d_ws,
                              size_t ws_size, hipStream_t stream) {
  const float* input_t = (const float*)d_in[0];
  const float* input_s = (const float*)d_in[1];
  const float* s2g_w   = (const float*)d_in[2];
  const float* s2g_b   = (const float*)d_in[3];
  const float* g2g_w   = (const float*)d_in[4];
  const float* g2g_b   = (const float*)d_in[5];
  const float* lnx_g   = (const float*)d_in[6];
  const float* lnx_b   = (const float*)d_in[7];
  const float* lny_g   = (const float*)d_in[8];
  const float* lny_b   = (const float*)d_in[9];
  const float* lnf_g   = (const float*)d_in[10];
  const float* lnf_b   = (const float*)d_in[11];
  const float* wq_w    = (const float*)d_in[12];
  const float* wq_b    = (const float*)d_in[13];
  const float* wk_w    = (const float*)d_in[14];
  const float* wk_b    = (const float*)d_in[15];
  const float* wv_w    = (const float*)d_in[16];
  const float* wv_b    = (const float*)d_in[17];
  const float* l1_w    = (const float*)d_in[18];
  const float* l1_b    = (const float*)d_in[19];
  const float* l2_w    = (const float*)d_in[20];
  const float* l2_b    = (const float*)d_in[21];
  const float* l3_w    = (const float*)d_in[22];
  const float* l3_b    = (const float*)d_in[23];

  float* gi   = (float*)d_ws;             // 3 * 12240
  float* xbuf = gi + 3 * GIN;             // 2 (ping-pong) * 3 * 4608

  prep_kernel<<<144, 256, 0, stream>>>(input_t, input_s, s2g_w, s2g_b, gi);
  gemv_kernel<<<432, 256, 0, stream>>>(g2g_w, g2g_b, gi, xbuf /* ver 0 */);

  int ver[3] = {0, 0, 0};
  const int pairs[3][2] = {{0, 1}, {2, 0}, {1, 2}};
  for (int layer = 0; layer < A_LAYERS; ++layer) {
    for (int p = 0; p < 3; ++p) {
      int m1 = pairs[p][0], m2 = pairs[p][1];
      const float* in1 = xbuf + ((size_t)ver[m1] * 3 + m1) * GROWS;
      const float* in2 = xbuf + ((size_t)ver[m2] * 3 + m2) * GROWS;
      float* o1 = xbuf + ((size_t)(1 - ver[m1]) * 3 + m1) * GROWS;
      float* o2 = xbuf + ((size_t)(1 - ver[m2]) * 3 + m2) * GROWS;
      attn_step_kernel<<<12, 256, 0, stream>>>(
          in1, in2, o1, o2, lnx_g, lnx_b, lny_g, lny_b, lnf_g, lnf_b,
          wq_w, wq_b, wk_w, wk_b, wv_w, wv_b, l1_w, l1_b, l2_w, l2_b,
          l3_w, l3_b, layer);
      ver[m1] ^= 1;
      ver[m2] ^= 1;
    }
  }
  pack_kernel<<<54, 256, 0, stream>>>(xbuf, ver[0], ver[1], ver[2],
                                      (float*)d_out);
}
